// GarmentDraper_86895778332920
// MI455X (gfx1250) — matrix operand order repaired and checked by reading, not compiled, not run on hardware
//
#include <hip/hip_runtime.h>
#include <stdint.h>

#define B_    2
#define VG_   20000
#define VB_   10475
#define FG_   40000
#define NQ    40000      /* B_*VG_ rows through the MLPs */
#define HID_  512
#define GFD_  256
#define EPS_  1e-8f
#define LNE_  1e-5f
#define MARG_ 0.005f

typedef __attribute__((ext_vector_type(16))) __bf16 v16bf;
typedef __attribute__((ext_vector_type(8)))  float  v8f;
typedef __attribute__((ext_vector_type(2)))  float  v2f;

static __device__ __forceinline__ unsigned short f2bf(float f) {
  unsigned int u = __float_as_uint(f);
  u += 0x7FFFu + ((u >> 16) & 1u);           // round-to-nearest-even
  return (unsigned short)(u >> 16);
}
static __device__ __forceinline__ float bf2f(unsigned short h) {
  return __uint_as_float(((unsigned int)h) << 16);
}

// ---------------------------------------------------------------------------
// Weight packing: row-major f32 W[din][dout] -> bf16-pair fragments laid out
// [ct16][ks32][lane 0..31][v 0..7] so each lane's 8 dwords per K-step are
// contiguous (32B -> two b128 loads). Matches V_WMMA 16-bit B layout:
// col = ct*16 + lane%16 ; K = ks*32 + (lane<16?0:8) + 2v + (v>=4?8:0) (+1 in high half).
// ---------------------------------------------------------------------------
__global__ void pack_w(const float* __restrict__ W, unsigned int* __restrict__ out,
                       int din, int dout) {
  const int nks = (din + 31) >> 5;
  const int nct = (dout + 15) >> 4;
  const size_t total = (size_t)nct * nks * 256;
  size_t idx = (size_t)blockIdx.x * blockDim.x + threadIdx.x;
  if (idx >= total) return;
  int v    = (int)(idx & 7);
  int lane = (int)((idx >> 3) & 31);
  size_t rest = idx >> 8;
  int ks = (int)(rest % nks);
  int ct = (int)(rest / nks);
  int col = ct * 16 + (lane & 15);
  int kb  = ks * 32 + ((lane < 16) ? 0 : 8) + 2 * v + ((v >= 4) ? 8 : 0);
  float lo = (col < dout && kb     < din) ? W[(size_t)kb * dout + col]       : 0.f;
  float hi = (col < dout && kb + 1 < din) ? W[(size_t)(kb + 1) * dout + col] : 0.f;
  out[idx] = (unsigned int)f2bf(lo) | ((unsigned int)f2bf(hi) << 16);
}

// ---------------------------------------------------------------------------
// bf16 WMMA GEMM: out = op(A[N,din] @ W[din,dout] + bias [+ res]).
// One wave -> 16 rows x 32 cols tile; K-loop step 32 with
// v_wmma_f32_16x16x32_bf16, fp32 accumulation. has1 hoisted out of the loop
// so each K-step is branch-free (no per-WMMA exec save/restore).
// ---------------------------------------------------------------------------
union ABfrag { v16bf v; uint4 q[2]; };

__global__ void __launch_bounds__(256) gemm_bf16(
    const unsigned short* __restrict__ A, int aw,
    const unsigned int*   __restrict__ Wp,
    const float*          __restrict__ bias,
    const unsigned short* __restrict__ resb,   // optional bf16 residual (stride dout)
    const float*          __restrict__ resf,   // optional f32 residual (stride dout)
    unsigned short*       __restrict__ outb,   // bf16 out (stride dout) or null
    float*                __restrict__ outf,   // f32  out (stride dout) if outb null
    int N, int din, int dout, int relu) {
  const int nks = (din + 31) >> 5;
  const int nct = (dout + 15) >> 4;
  const int ncp = (nct + 1) >> 1;
  const int lane = threadIdx.x & 31;
  const int wave = (int)((blockIdx.x * blockDim.x + threadIdx.x) >> 5);
  const int rowTiles = N >> 4;
  if (wave >= rowTiles * ncp) return;
  const int rt = wave / ncp;
  const int cp = wave - rt * ncp;
  const int ct0 = cp << 1;
  const bool has1 = (ct0 + 1) < nct;

  const int row  = (rt << 4) + (lane & 15);
  const int kSel = (lane < 16) ? 0 : 8;       // A half-wave K offset
  const unsigned short* arow = A + (size_t)row * aw;
  const unsigned int* bbase0 = Wp + (((size_t)ct0 * nks) << 8) + ((size_t)lane << 3);
  const unsigned int* bbase1 = bbase0 + ((size_t)nks << 8);
  __builtin_prefetch(arow, 0, 1);
  __builtin_prefetch(bbase0, 0, 1);

  v8f acc0 = {0.f, 0.f, 0.f, 0.f, 0.f, 0.f, 0.f, 0.f};
  v8f acc1 = acc0;
  if (has1) {
    for (int ks = 0; ks < nks; ++ks) {
      ABfrag a, b0, b1;
      const unsigned short* ap = arow + (ks << 5) + kSel;
      a.q[0] = *reinterpret_cast<const uint4*>(ap);        // K {0..7} / {8..15}
      a.q[1] = *reinterpret_cast<const uint4*>(ap + 16);   // K {16..23} / {24..31}
      const unsigned int* bp = bbase0 + ((size_t)ks << 8);
      b0.q[0] = *reinterpret_cast<const uint4*>(bp);
      b0.q[1] = *reinterpret_cast<const uint4*>(bp + 4);
      const unsigned int* bp1 = bbase1 + ((size_t)ks << 8);
      b1.q[0] = *reinterpret_cast<const uint4*>(bp1);
      b1.q[1] = *reinterpret_cast<const uint4*>(bp1 + 4);
      acc0 = __builtin_amdgcn_wmma_f32_16x16x32_bf16(false, a.v, false, b0.v,
                                                     (short)0, acc0, false, false);
      acc1 = __builtin_amdgcn_wmma_f32_16x16x32_bf16(false, a.v, false, b1.v,
                                                     (short)0, acc1, false, false);
    }
  } else {
    for (int ks = 0; ks < nks; ++ks) {
      ABfrag a, b0;
      const unsigned short* ap = arow + (ks << 5) + kSel;
      a.q[0] = *reinterpret_cast<const uint4*>(ap);
      a.q[1] = *reinterpret_cast<const uint4*>(ap + 16);
      const unsigned int* bp = bbase0 + ((size_t)ks << 8);
      b0.q[0] = *reinterpret_cast<const uint4*>(bp);
      b0.q[1] = *reinterpret_cast<const uint4*>(bp + 4);
      acc0 = __builtin_amdgcn_wmma_f32_16x16x32_bf16(false, a.v, false, b0.v,
                                                     (short)0, acc0, false, false);
    }
  }
  // C layout: lane holds col = lane%16; VGPR r -> row r + 8*(lane/16)
  const int mBase = (rt << 4) + ((lane >> 4) << 3);
  const int nc = lane & 15;
  for (int t = 0; t < 2; ++t) {
    if (t == 1 && !has1) break;
    const int col = ((ct0 + t) << 4) + nc;
    if (col >= dout) continue;
    const float bv = bias ? bias[col] : 0.f;
    v8f acc = t ? acc1 : acc0;
#pragma unroll
    for (int r = 0; r < 8; ++r) {
      const size_t idx = (size_t)(mBase + r) * dout + col;
      float v = acc[r] + bv;
      if (resb) v += bf2f(resb[idx]);
      if (resf) v += resf[idx];
      if (relu) v = fmaxf(v, 0.f);
      if (outb) outb[idx] = f2bf(v);
      else      outf[idx] = v;
    }
  }
}

// ---------------------------------------------------------------------------
// Nearest neighbor via V_WMMA_F32_16X16X4_F32 (full f32 precision):
//   A row m  = [-2qx, -2qy, -2qz, 1]   (16x4, loop-invariant)
//   B col n  = [ rx,   ry,   rz, |r|^2] (4x16 per tile)
//   C[m][n]  = |r_n|^2 - 2 q_m . r_n   (argmin-equivalent to d2; |q|^2 added
//                                       only for mind)
// Per-lane running (min, argmin) over its column, then shfl_xor butterfly in
// each 16-lane half; ties keep the lowest index (jnp.argmin semantics).
// One wave per 16-query tile; nq must be a multiple of 16 (20000 = 1250*16).
// ---------------------------------------------------------------------------
__global__ void __launch_bounds__(256) nn_wmma(
    const float* __restrict__ q, const float* __restrict__ ref,
    int nq, int nref, float* __restrict__ nb, float* __restrict__ mind,
    unsigned short* __restrict__ cat6) {
#if defined(__gfx1250__)
  const int lane = threadIdx.x & 31;
  const int wave = (int)((blockIdx.x * blockDim.x + threadIdx.x) >> 5);
  const int rowTiles = nq >> 4;
  if (wave >= B_ * rowTiles) return;             // wave-uniform exit
  const int b  = wave / rowTiles;
  const int rt = wave - b * rowTiles;
  const float* Q = q   + (size_t)b * nq * 3;
  const float* R = ref + (size_t)b * nref * 3;

  const int m   = lane & 15;
  const int row = rt * 16 + m;
  const float qx = Q[row*3], qy = Q[row*3+1], qz = Q[row*3+2];
  v2f a;
  a[0] = (lane < 16) ? (-2.f * qx) : (-2.f * qz);   // K0 / K2
  a[1] = (lane < 16) ? (-2.f * qy) : 1.f;           // K1 / K3

  float best[8];
  int   bidx[8];
#pragma unroll
  for (int r = 0; r < 8; ++r) { best[r] = 3.0e38f; bidx[r] = 0; }
  const v8f cz = {0.f, 0.f, 0.f, 0.f, 0.f, 0.f, 0.f, 0.f};

  const int ntiles = (nref + 15) >> 4;
  for (int ct = 0; ct < ntiles; ++ct) {
    const int n = ct * 16 + m;
    const bool valid = n < nref;
    const int nc = valid ? n : (nref - 1);
    float rx = R[nc*3], ry = R[nc*3+1], rz = R[nc*3+2];
    float r2 = rx*rx + ry*ry + rz*rz;
    if (!valid) { rx = 0.f; ry = 0.f; rz = 0.f; r2 = 1.0e30f; }
    v2f bf;
    bf[0] = (lane < 16) ? rx : rz;                  // K0 / K2
    bf[1] = (lane < 16) ? ry : r2;                  // K1 / K3
    v8f c = __builtin_amdgcn_wmma_f32_16x16x4_f32(false, a, false, bf,
                                                  (short)0, cz, false, false);
#pragma unroll
    for (int r = 0; r < 8; ++r) {
      if (c[r] < best[r]) { best[r] = c[r]; bidx[r] = n; }  // strict <: first idx
    }
  }
  // reduce the 16 lanes of each half (rows r / r+8), lowest index on ties
#pragma unroll
  for (int r = 0; r < 8; ++r) {
    float bv = best[r]; int bix = bidx[r];
    for (int s = 1; s < 16; s <<= 1) {
      float ov = __shfl_xor(bv, s, 32);
      int   oi = __shfl_xor(bix, s, 32);
      if (ov < bv || (ov == bv && oi < bix)) { bv = ov; bix = oi; }
    }
    best[r] = bv; bidx[r] = bix;
  }
  if ((lane & 15) == 0) {                          // lane 0 -> rows 0..7, lane 16 -> rows 8..15
    const int mrow = rt * 16 + ((lane >> 4) << 3);
#pragma unroll
    for (int r = 0; r < 8; ++r) {
      const int gr = mrow + r;
      const int bi = bidx[r];
      const float nx = R[bi*3], ny = R[bi*3+1], nz = R[bi*3+2];
      const size_t o = ((size_t)b * nq + gr) * 3;
      nb[o] = nx; nb[o+1] = ny; nb[o+2] = nz;
      const float gx = Q[gr*3], gy = Q[gr*3+1], gz = Q[gr*3+2];
      if (mind) {
        float d2 = best[r] + gx*gx + gy*gy + gz*gz;
        mind[(size_t)b * nq + gr] = sqrtf(fmaxf(d2, 0.f));
      }
      if (cat6) {
        unsigned short* cc = cat6 + ((size_t)b * nq + gr) * 32;
        cc[0] = f2bf(gx); cc[1] = f2bf(gy); cc[2] = f2bf(gz);
        cc[3] = f2bf(nx); cc[4] = f2bf(ny); cc[5] = f2bf(nz);
        for (int j = 6; j < 32; ++j) cc[j] = 0;
      }
    }
  }
#endif
}

// ---------------------------------------------------------------------------
// VALU+LDS fallback NN (kept compiled, not launched).
// ---------------------------------------------------------------------------
#define NN_TILE 1024
__global__ void __launch_bounds__(256) nn_valu(
    const float* __restrict__ q, const float* __restrict__ ref,
    int nq, int nref, float* __restrict__ nb, float* __restrict__ mind,
    unsigned short* __restrict__ cat6) {
  __shared__ float tile[NN_TILE * 3];
  const int b = blockIdx.y;
  const int i = blockIdx.x * blockDim.x + threadIdx.x;
  const float* Q = q   + (size_t)b * nq * 3;
  const float* R = ref + (size_t)b * nref * 3;
  float qx = 0.f, qy = 0.f, qz = 0.f;
  const bool active = i < nq;
  if (active) { qx = Q[i*3]; qy = Q[i*3+1]; qz = Q[i*3+2]; }
  float best = 3.4e38f;
  int bi = 0;
  for (int t0 = 0; t0 < nref; t0 += NN_TILE) {
    int cnt = nref - t0; if (cnt > NN_TILE) cnt = NN_TILE;
    __syncthreads();
    for (int j = threadIdx.x; j < cnt * 3; j += blockDim.x) tile[j] = R[t0*3 + j];
    __syncthreads();
    for (int j = 0; j < cnt; ++j) {
      float dx = qx - tile[j*3], dy = qy - tile[j*3+1], dz = qz - tile[j*3+2];
      float d2 = dx*dx + dy*dy + dz*dz;
      if (d2 < best) { best = d2; bi = t0 + j; }
    }
  }
  if (!active) return;
  const float nx = R[bi*3], ny = R[bi*3+1], nz = R[bi*3+2];
  const size_t o = ((size_t)b * nq + i) * 3;
  nb[o] = nx; nb[o+1] = ny; nb[o+2] = nz;
  if (mind) mind[(size_t)b * nq + i] = sqrtf(fmaxf(best, 0.f));
  if (cat6) {
    unsigned short* c = cat6 + ((size_t)b * nq + i) * 32;
    c[0] = f2bf(qx); c[1] = f2bf(qy); c[2] = f2bf(qz);
    c[3] = f2bf(nx); c[4] = f2bf(ny); c[5] = f2bf(nz);
    for (int j = 6; j < 32; ++j) c[j] = 0;
  }
}

// One wave per 512-wide row: relu(layernorm(x)) in place (bf16 storage).
__global__ void __launch_bounds__(256) ln_relu_512(
    unsigned short* __restrict__ H, const float* __restrict__ g,
    const float* __restrict__ bb, int N) {
  const int wave = (int)((blockIdx.x * blockDim.x + threadIdx.x) >> 5);
  const int lane = threadIdx.x & 31;
  if (wave >= N) return;
  unsigned short* row = H + (size_t)wave * 512;
  float vals[16], s = 0.f, s2 = 0.f;
#pragma unroll
  for (int i = 0; i < 16; ++i) {
    float v = bf2f(row[lane + i * 32]);
    vals[i] = v; s += v; s2 += v * v;
  }
  for (int off = 16; off >= 1; off >>= 1) {
    s  += __shfl_xor(s,  off, 32);
    s2 += __shfl_xor(s2, off, 32);
  }
  const float mean = s * (1.f / 512.f);
  const float var  = s2 * (1.f / 512.f) - mean * mean;
  const float inv  = rsqrtf(var + LNE_);
#pragma unroll
  for (int i = 0; i < 16; ++i) {
    const int c = lane + i * 32;
    float v = (vals[i] - mean) * inv * g[c] + bb[c];
    row[c] = f2bf(fmaxf(v, 0.f));
  }
}

// Tiny material MLP: (2,8) -> 64 relu -> 256 (f32, one block).
__global__ void mat_mlp(const float* __restrict__ mp,
                        const float* __restrict__ w0, const float* __restrict__ b0,
                        const float* __restrict__ w1, const float* __restrict__ b1,
                        float* __restrict__ mf) {
  __shared__ float h[2][64];
  const int t = threadIdx.x;
  if (t < 128) {
    int b = t >> 6, j = t & 63;
    float a = b0[j];
    for (int k = 0; k < 8; ++k) a += mp[b*8 + k] * w0[k*64 + j];
    h[b][j] = fmaxf(a, 0.f);
  }
  __syncthreads();
  for (int o = t; o < 2 * 256; o += blockDim.x) {
    int b = o >> 8, j = o & 255;
    float a = b1[j];
    for (int k = 0; k < 64; ++k) a += h[b][k] * w1[k*256 + j];
    mf[b*256 + j] = a;
  }
}

// x0[row][0:3]=coarse, [3:259]=mf[batch], [259:288]=0 (bf16, stride 288)
__global__ void build_x0(const float* __restrict__ coarse, const float* __restrict__ mf,
                         unsigned short* __restrict__ x0, int nq) {
  size_t idx = (size_t)blockIdx.x * blockDim.x + threadIdx.x;
  if (idx >= (size_t)2 * nq * 288) return;
  int j = (int)(idx % 288);
  size_t row = idx / 288;
  int b = (int)(row / nq);
  float v = 0.f;
  if (j < 3)        v = coarse[row * 3 + j];
  else if (j < 259) v = mf[b * 256 + (j - 3)];
  x0[idx] = f2bf(v);
}

__global__ void ln3_relu(float* __restrict__ h, const float* __restrict__ g,
                         const float* __restrict__ bb, int n) {
  int i = blockIdx.x * blockDim.x + threadIdx.x;
  if (i >= n) return;
  float x0 = h[i*3], x1 = h[i*3+1], x2 = h[i*3+2];
  float m = (x0 + x1 + x2) * (1.f / 3.f);
  float v = ((x0-m)*(x0-m) + (x1-m)*(x1-m) + (x2-m)*(x2-m)) * (1.f / 3.f);
  float inv = rsqrtf(v + LNE_);
  h[i*3]   = fmaxf((x0 - m) * inv * g[0] + bb[0], 0.f);
  h[i*3+1] = fmaxf((x1 - m) * inv * g[1] + bb[1], 0.f);
  h[i*3+2] = fmaxf((x2 - m) * inv * g[2] + bb[2], 0.f);
}

// offsets = h@w2(3x3) + b2 + skip ; draped0 = coarse + 0.05*offsets
__global__ void refine3_final(const float* __restrict__ h, const float* __restrict__ w2,
                              const float* __restrict__ b2, const float* __restrict__ sres,
                              const float* __restrict__ coarse,
                              float* __restrict__ offs, float* __restrict__ draped0, int n) {
  int i = blockIdx.x * blockDim.x + threadIdx.x;
  if (i >= n) return;
  float a0 = h[i*3], a1 = h[i*3+1], a2 = h[i*3+2];
#pragma unroll
  for (int k = 0; k < 3; ++k) {
    float o = a0 * w2[0*3+k] + a1 * w2[1*3+k] + a2 * w2[2*3+k] + b2[k] + sres[i*3+k];
    offs[i*3+k] = o;
    draped0[i*3+k] = coarse[i*3+k] + 0.05f * o;
  }
}

__global__ void pushback(const float* __restrict__ d0, const float* __restrict__ nb2,
                         const float* __restrict__ mind2, float* __restrict__ out, int n) {
  int i = blockIdx.x * blockDim.x + threadIdx.x;
  if (i >= n) return;
  float dx = d0[i*3] - nb2[i*3], dy = d0[i*3+1] - nb2[i*3+1], dz = d0[i*3+2] - nb2[i*3+2];
  float inv = 1.f / (sqrtf(dx*dx + dy*dy + dz*dz) + EPS_);
  float tc = (mind2[i] < MARG_) ? 1.f : 0.f;
  out[i*3]   = d0[i*3]   + dx * inv * MARG_ * tc;
  out[i*3+1] = d0[i*3+1] + dy * inv * MARG_ * tc;
  out[i*3+2] = d0[i*3+2] + dz * inv * MARG_ * tc;
}

__global__ void zero_f32(float* p, int n) {
  int i = blockIdx.x * blockDim.x + threadIdx.x;
  if (i < n) p[i] = 0.f;
}

__global__ void face_normals(const float* __restrict__ draped, const int* __restrict__ faces,
                             float* __restrict__ vn, int nf, int nv) {
  int idx = blockIdx.x * blockDim.x + threadIdx.x;
  if (idx >= 2 * nf) return;
  int b = idx / nf, f = idx - b * nf;
  const float* D = draped + (size_t)b * nv * 3;
  int i0 = faces[f*3], i1 = faces[f*3+1], i2 = faces[f*3+2];
  float ax = D[i0*3], ay = D[i0*3+1], az = D[i0*3+2];
  float e1x = D[i1*3]-ax, e1y = D[i1*3+1]-ay, e1z = D[i1*3+2]-az;
  float e2x = D[i2*3]-ax, e2y = D[i2*3+1]-ay, e2z = D[i2*3+2]-az;
  float nx = e1y*e2z - e1z*e2y, ny = e1z*e2x - e1x*e2z, nz = e1x*e2y - e1y*e2x;
  float inv = 1.f / (sqrtf(nx*nx + ny*ny + nz*nz) + EPS_);
  nx *= inv; ny *= inv; nz *= inv;
  float* V = vn + (size_t)b * nv * 3;
  atomicAdd(&V[i0*3], nx); atomicAdd(&V[i0*3+1], ny); atomicAdd(&V[i0*3+2], nz);
  atomicAdd(&V[i1*3], nx); atomicAdd(&V[i1*3+1], ny); atomicAdd(&V[i1*3+2], nz);
  atomicAdd(&V[i2*3], nx); atomicAdd(&V[i2*3+1], ny); atomicAdd(&V[i2*3+2], nz);
}

__global__ void normalize_vn(float* __restrict__ vn, int n) {
  int i = blockIdx.x * blockDim.x + threadIdx.x;
  if (i >= n) return;
  float x = vn[i*3], y = vn[i*3+1], z = vn[i*3+2];
  float inv = 1.f / (sqrtf(x*x + y*y + z*z) + EPS_);
  vn[i*3] = x * inv; vn[i*3+1] = y * inv; vn[i*3+2] = z * inv;
}

// ---------------------------------------------------------------------------
extern "C" void kernel_launch(void* const* d_in, const int* in_sizes, int n_in,
                              void* d_out, int out_size, void* d_ws, size_t ws_size,
                              hipStream_t stream) {
  (void)out_size; (void)ws_size;
  if (n_in < 43) return;

  // --- resolve input indices (handle sorted-vs-insertion pytree order) ---
  int gv, gf, bvx, mp;
  if (in_sizes[0] == 62724) { bvx = 1; gf = 2; gv = 3; mp = 4; }   // sorted top-level
  else                      { gv = 0; gf = 1; bvx = 2; mp = 4; }   // insertion order
  const int base = 5;
  const int c0w = base+0, c0b = base+1, c1w = base+2, c1b = base+3, c2w = base+4, c2b = base+5;
  const int m0w = base+6, m0b = base+7, m1w = base+8, m1b = base+9;
  const int r0 = base + 10;
  const bool psorted = (in_sizes[r0] == 512);  // ln_b first if dict keys sorted
  // R[blk] = {lnb, lng, skip_w, skip_b, w1, b1, w2, b2}
  int R[4][8];
  const int starts[4] = { r0, r0 + 8, r0 + 14, r0 + 20 };
  for (int blk = 0; blk < 4; ++blk) {
    int s = starts[blk];
    bool hs = (blk == 0 || blk == 3);
    if (psorted) {
      if (hs) { R[blk][0]=s+0; R[blk][1]=s+1; R[blk][2]=s+2; R[blk][3]=s+3;
                R[blk][4]=s+4; R[blk][5]=s+5; R[blk][6]=s+6; R[blk][7]=s+7; }
      else    { R[blk][0]=s+0; R[blk][1]=s+1; R[blk][2]=-1;  R[blk][3]=-1;
                R[blk][4]=s+2; R[blk][5]=s+3; R[blk][6]=s+4; R[blk][7]=s+5; }
    } else {  // insertion order: w1, ln_g, ln_b, w2, skip
      if (hs) { R[blk][4]=s+0; R[blk][5]=s+1; R[blk][1]=s+2; R[blk][0]=s+3;
                R[blk][6]=s+4; R[blk][7]=s+5; R[blk][2]=s+6; R[blk][3]=s+7; }
      else    { R[blk][4]=s+0; R[blk][5]=s+1; R[blk][1]=s+2; R[blk][0]=s+3;
                R[blk][6]=s+4; R[blk][7]=s+5; R[blk][2]=-1;  R[blk][3]=-1; }
    }
  }
  auto F = [&](int i) { return (const float*)d_in[i]; };

  // --- workspace carve-out ---
  char* ws = (char*)d_ws;
  size_t off = 0;
  auto alloc = [&](size_t bytes) -> void* {
    void* p = ws + off;
    off += (bytes + 255) & ~(size_t)255;
    return p;
  };
  float* nb1     = (float*)alloc((size_t)NQ * 3 * 4);
  float* coarse  = (float*)alloc((size_t)NQ * 3 * 4);
  float* nb2     = (float*)alloc((size_t)NQ * 3 * 4);
  float* mind2   = (float*)alloc((size_t)NQ * 4);
  float* mf      = (float*)alloc((size_t)B_ * GFD_ * 4);
  float* h3a     = (float*)alloc((size_t)NQ * 3 * 4);
  float* sres3   = (float*)alloc((size_t)NQ * 3 * 4);
  float* draped0 = (float*)alloc((size_t)NQ * 3 * 4);
  unsigned short* cat6 = (unsigned short*)alloc((size_t)NQ * 32 * 2);
  unsigned short* aH   = (unsigned short*)alloc((size_t)NQ * 512 * 2);
  unsigned short* aH2  = (unsigned short*)alloc((size_t)NQ * 512 * 2);
  unsigned short* aXC  = (unsigned short*)alloc((size_t)NQ * 512 * 2);
  unsigned short* x0   = (unsigned short*)alloc((size_t)NQ * 288 * 2);
  auto pwbytes = [](int din, int dout) -> size_t {
    return (size_t)((dout + 15) / 16) * ((din + 31) / 32) * 256 * 4;
  };
  unsigned int* pC0  = (unsigned int*)alloc(pwbytes(6, 512));
  unsigned int* pC1  = (unsigned int*)alloc(pwbytes(512, 512));
  unsigned int* pC2  = (unsigned int*)alloc(pwbytes(512, 3));
  unsigned int* p0s  = (unsigned int*)alloc(pwbytes(259, 512));
  unsigned int* p0w1 = (unsigned int*)alloc(pwbytes(259, 512));
  unsigned int* p0w2 = (unsigned int*)alloc(pwbytes(512, 512));
  unsigned int* p1w1 = (unsigned int*)alloc(pwbytes(512, 512));
  unsigned int* p1w2 = (unsigned int*)alloc(pwbytes(512, 512));
  unsigned int* p2w1 = (unsigned int*)alloc(pwbytes(512, 512));
  unsigned int* p2w2 = (unsigned int*)alloc(pwbytes(512, 512));
  unsigned int* p3s  = (unsigned int*)alloc(pwbytes(512, 3));
  unsigned int* p3w1 = (unsigned int*)alloc(pwbytes(512, 3));

  // --- pack all GEMM weights to bf16 fragments ---
  auto pack = [&](int widx, unsigned int* dst, int din, int dout) {
    size_t total = (size_t)((dout + 15) / 16) * ((din + 31) / 32) * 256;
    pack_w<<<dim3((unsigned)((total + 255) / 256)), dim3(256), 0, stream>>>(F(widx), dst, din, dout);
  };
  pack(c0w, pC0, 6, 512);      pack(c1w, pC1, 512, 512);  pack(c2w, pC2, 512, 3);
  pack(R[0][2], p0s, 259, 512); pack(R[0][4], p0w1, 259, 512); pack(R[0][6], p0w2, 512, 512);
  pack(R[1][4], p1w1, 512, 512); pack(R[1][6], p1w2, 512, 512);
  pack(R[2][4], p2w1, 512, 512); pack(R[2][6], p2w2, 512, 512);
  pack(R[3][2], p3s, 512, 3);   pack(R[3][4], p3w1, 512, 3);

  auto gemm = [&](const unsigned short* A, int aw, const unsigned int* Wp, int bidx,
                  const unsigned short* rb, const float* rf,
                  unsigned short* ob, float* of, int din, int dout, int relu) {
    int nct = (dout + 15) / 16, ncp = (nct + 1) / 2;
    int waves = (NQ / 16) * ncp;
    gemm_bf16<<<dim3((unsigned)((waves + 7) / 8)), dim3(256), 0, stream>>>(
        A, aw, Wp, F(bidx), rb, rf, ob, of, NQ, din, dout, relu);
  };
  auto ln = [&](unsigned short* H, int gi, int bi) {
    ln_relu_512<<<dim3((NQ + 7) / 8), dim3(256), 0, stream>>>(H, F(gi), F(bi), NQ);
  };
  auto nn = [&](const float* qp, float* nbp, float* mindp, unsigned short* catp) {
    int waves = B_ * (VG_ / 16);
    nn_wmma<<<dim3((waves * 32 + 255) / 256), dim3(256), 0, stream>>>(
        qp, F(bvx), VG_, VB_, nbp, mindp, catp);
  };

  float* out_draped  = (float*)d_out;
  float* out_offsets = out_draped + (size_t)NQ * 3;
  float* out_vn      = out_offsets + (size_t)NQ * 3;

  // 1) nearest body vertex + concat6
  nn(F(gv), nb1, nullptr, cat6);
  // 2) corr MLP -> coarse = nb + lin(h)
  gemm(cat6, 32, pC0, c0b, nullptr, nullptr, aH,  nullptr, 6,   512, 1);
  gemm(aH,  512, pC1, c1b, nullptr, nullptr, aH2, nullptr, 512, 512, 1);
  gemm(aH2, 512, pC2, c2b, nullptr, nb1,     nullptr, coarse, 512, 3, 0);
  // 3) material features + refine input
  mat_mlp<<<dim3(1), dim3(256), 0, stream>>>(F(mp), F(m0w), F(m0b), F(m1w), F(m1b), mf);
  {
    size_t total = (size_t)NQ * 288;
    build_x0<<<dim3((unsigned)((total + 255) / 256)), dim3(256), 0, stream>>>(coarse, mf, x0, VG_);
  }
  // 4) refine block 0 (259 -> 512, projected skip)
  gemm(x0, 288, p0s,  R[0][3], nullptr, nullptr, aXC, nullptr, 259, 512, 0);
  gemm(x0, 288, p0w1, R[0][5], nullptr, nullptr, aH,  nullptr, 259, 512, 0);
  ln(aH, R[0][1], R[0][0]);
  gemm(aH, 512, p0w2, R[0][7], aXC, nullptr, aXC, nullptr, 512, 512, 0);
  // 5) refine block 1 (identity skip)
  gemm(aXC, 512, p1w1, R[1][5], nullptr, nullptr, aH, nullptr, 512, 512, 0);
  ln(aH, R[1][1], R[1][0]);
  gemm(aH, 512, p1w2, R[1][7], aXC, nullptr, aH2, nullptr, 512, 512, 0);
  // 6) refine block 2 (identity skip)
  gemm(aH2, 512, p2w1, R[2][5], nullptr, nullptr, aH, nullptr, 512, 512, 0);
  ln(aH, R[2][1], R[2][0]);
  gemm(aH, 512, p2w2, R[2][7], aH2, nullptr, aXC, nullptr, 512, 512, 0);
  // 7) refine block 3 (512 -> 3)
  gemm(aXC, 512, p3w1, R[3][5], nullptr, nullptr, nullptr, h3a, 512, 3, 0);
  ln3_relu<<<dim3((NQ + 255) / 256), dim3(256), 0, stream>>>(h3a, F(R[3][1]), F(R[3][0]), NQ);
  gemm(aXC, 512, p3s, R[3][3], nullptr, nullptr, nullptr, sres3, 512, 3, 0);
  refine3_final<<<dim3((NQ + 255) / 256), dim3(256), 0, stream>>>(
      h3a, F(R[3][6]), F(R[3][7]), sres3, coarse, out_offsets, draped0, NQ);
  // 8) second NN + margin pushback
  nn(draped0, nb2, mind2, nullptr);
  pushback<<<dim3((NQ + 255) / 256), dim3(256), 0, stream>>>(draped0, nb2, mind2, out_draped, NQ);
  // 9) vertex normals
  zero_f32<<<dim3((NQ * 3 + 255) / 256), dim3(256), 0, stream>>>(out_vn, NQ * 3);
  face_normals<<<dim3((B_ * FG_ + 255) / 256), dim3(256), 0, stream>>>(
      out_draped, (const int*)d_in[gf], out_vn, FG_, VG_);
  normalize_vn<<<dim3((NQ + 255) / 256), dim3(256), 0, stream>>>(out_vn, NQ);
}
